// Conv2d_int8_est_T_50483045597926
// MI455X (gfx1250) — 8-bit matrix operand order repaired and checked by reading, compiled, not run on hardware
//
#include <hip/hip_runtime.h>

typedef __attribute__((ext_vector_type(8))) int v8i;
typedef __attribute__((ext_vector_type(4))) int v4i;
typedef __attribute__((ext_vector_type(2))) int v2i;

#define B_    8
#define CIN   64
#define COUT  128
#define HH    32
#define WW    32
#define KDIM  576
#define NPIX  8192

#define WS_WQ_OFF   64
#define WS_COL_OFF  73792

__global__ void init_ws(unsigned int* ws) {
    if (threadIdx.x < 2) ws[threadIdx.x] = 0u;
}

__global__ void absmax_kernel(const float* __restrict__ src, int n, unsigned int* dst) {
    __shared__ unsigned int red[256];
    unsigned int m = 0;
    for (int i = blockIdx.x * blockDim.x + threadIdx.x; i < n; i += gridDim.x * blockDim.x) {
        unsigned int v = __float_as_uint(fabsf(src[i]));
        m = v > m ? v : m;
    }
    red[threadIdx.x] = m;
    __syncthreads();
    for (int s = 128; s > 0; s >>= 1) {
        if ((int)threadIdx.x < s) {
            unsigned int o = red[threadIdx.x + s];
            if (o > red[threadIdx.x]) red[threadIdx.x] = o;
        }
        __syncthreads();
    }
    if (threadIdx.x == 0) atomicMax(dst, red[0]);
}

__global__ void compute_scales(const unsigned int* __restrict__ am,
                               const float* __restrict__ Tf_in,
                               const float* __restrict__ Tw_in,
                               float* __restrict__ scal) {
    if (threadIdx.x == 0) {
        float Tf = 0.95f * Tf_in[0] + 0.05f * __uint_as_float(am[0]);
        float Tw = 0.95f * Tw_in[0] + 0.05f * __uint_as_float(am[1]);
        scal[0] = 127.0f / Tf;
        scal[1] = 127.0f / Tw;
        scal[2] = (Tf / 127.0f) * (Tw / 127.0f);
    }
}

__device__ __forceinline__ signed char quant(float v, float s) {
    float q = rintf(v * s);
    q = fminf(fmaxf(q, -128.0f), 127.0f);
    return (signed char)(int)q;
}

__global__ void quantize_w(const float* __restrict__ w, const float* __restrict__ scal,
                           signed char* __restrict__ wq) {
    int i = blockIdx.x * blockDim.x + threadIdx.x;
    if (i < COUT * KDIM) wq[i] = quant(w[i], scal[1]);
}

__global__ void build_col(const float* __restrict__ x, const float* __restrict__ scal,
                          signed char* __restrict__ col) {
    int i = blockIdx.x * blockDim.x + threadIdx.x;
    if (i >= NPIX * KDIM) return;
    int p  = i / KDIM,  k  = i - p * KDIM;
    int ci = k / 9,     r  = k - ci * 9;
    int ki = r / 3,     kj = r - ki * 3;
    int b  = p >> 10,   hw = p & 1023;
    int oh = hw >> 5,   ow = hw & 31;
    int ih = oh + ki - 1, iw = ow + kj - 1;
    signed char q = 0;
    if ((unsigned)ih < 32u && (unsigned)iw < 32u)
        q = quant(x[((b * CIN + ci) * HH + ih) * WW + iw], scal[0]);
    col[i] = q;
}

__global__ void __launch_bounds__(256)
gemm_iu8(const signed char* __restrict__ wq, const signed char* __restrict__ col,
         const float* __restrict__ scal, const float* __restrict__ bias,
         float* __restrict__ out) {
    int wave = ((int)blockIdx.x * 256 + (int)threadIdx.x) >> 5;
    int lane = threadIdx.x & 31;
    int tn   = wave & 7;
    int tm   = wave >> 3;
    int hf   = lane >> 4;
    int l16  = lane & 15;

    const signed char* arow = wq  + (tn * 16 + l16) * KDIM;
    const signed char* brow = col + (size_t)(tm * 16 + l16) * KDIM;

    v8i acc = {};
#pragma unroll
    for (int kk = 0; kk < KDIM; kk += 64) {
        v2i a01 = *(const v2i*)(arow + kk +      hf * 16);
        v2i a23 = *(const v2i*)(arow + kk +  8 + hf * 16);
        v2i a45 = *(const v2i*)(arow + kk + 32 + hf * 16);
        v2i a67 = *(const v2i*)(arow + kk + 40 + hf * 16);
        v4i b03 = *(const v4i*)(brow + kk +      hf * 16);
        v4i b47 = *(const v4i*)(brow + kk + 32 + hf * 16);
        v8i Am = { a01.x, a01.y, a23.x, a23.y, a45.x, a45.y, a67.x, a67.y };
        v8i Bm = { b03.x, b03.y, b03.z, b03.w, b47.x, b47.y, b47.z, b47.w };
        acc = __builtin_amdgcn_wmma_i32_16x16x64_iu8(true, Am, true, Bm, acc, false, false);
    }

    float dq  = scal[2];
    int   pix = tm * 16 + l16;
    int   b   = pix >> 10, hw = pix & 1023;
#pragma unroll
    for (int r = 0; r < 8; ++r) {
        int co = tn * 16 + hf * 8 + r;
        out[((size_t)(b * COUT + co) << 10) + hw] = (float)acc[r] * dq + bias[co];
    }
}

extern "C" void kernel_launch(void* const* d_in, const int* in_sizes, int n_in,
                              void* d_out, int out_size, void* d_ws, size_t ws_size,
                              hipStream_t stream) {
    const float* x    = (const float*)d_in[0];
    const float* w    = (const float*)d_in[1];
    const float* bias = (const float*)d_in[2];
    const float* Tf   = (const float*)d_in[5];
    const float* Tw   = (const float*)d_in[6];

    unsigned int* am   = (unsigned int*)d_ws;
    float*        scal = (float*)((char*)d_ws + 16);
    signed char*  wq   = (signed char*)d_ws + WS_WQ_OFF;
    signed char*  col  = (signed char*)d_ws + WS_COL_OFF;
    float*        out  = (float*)d_out;

    init_ws<<<1, 64, 0, stream>>>(am);
    absmax_kernel<<<256, 256, 0, stream>>>(x, B_ * CIN * HH * WW, am);
    absmax_kernel<<<64, 256, 0, stream>>>(w, COUT * KDIM, am + 1);
    compute_scales<<<1, 32, 0, stream>>>(am, Tf, Tw, scal);
    quantize_w<<<(COUT * KDIM + 255) / 256, 256, 0, stream>>>(w, scal, wq);
    build_col<<<(NPIX * KDIM + 255) / 256, 256, 0, stream>>>(x, scal, col);
    gemm_iu8<<<512, 256, 0, stream>>>(wq, col, scal, bias, out);
}
